// MultiHeadAttention_74990128988667
// MI455X (gfx1250) — matrix operand order repaired and checked by reading, not compiled, not run on hardware
//
#include <hip/hip_runtime.h>
#include <hip/hip_bf16.h>

// ---------------- problem constants ----------------
#define EMBED 768
#define NHEAD 12
#define HDIM 64
#define BATCH 2
#define NTOK 3073
#define NPAD 3104            // 3073 padded to multiple of 32
#define MROWS (BATCH * NPAD) // 6208 padded rows total
#define QKVDIM (3 * EMBED)   // 2304
#define SCALE 0.125f         // 64^-0.5
#define NQT (NPAD / 16)      // 194 query tiles per (b,h)
#define NKB (NPAD / 32)      // 97 key blocks of 32

typedef unsigned short u16;
typedef unsigned int u32;
typedef unsigned long long u64;
typedef __attribute__((ext_vector_type(16))) __bf16 v16bf;
typedef __attribute__((ext_vector_type(8))) float v8f;
typedef __attribute__((ext_vector_type(4))) u32 u32x4;
typedef __attribute__((ext_vector_type(8))) u32 u32x8;

union FragU {
    v16bf v;
    uint4 q[2];
};

// float -> bf16 round-to-nearest-even
__device__ __forceinline__ u16 f2bf(float f) {
    unsigned u = __float_as_uint(f);
    unsigned r = u + 0x7FFFu + ((u >> 16) & 1u);
    return (u16)(r >> 16);
}
__device__ __forceinline__ float bf2f(u16 s) {
    return __uint_as_float(((unsigned)s) << 16);
}

// A fragment: 16x32 bf16, row m = lane&15; K-chunks {0..7,16..23} lanes 0-15,
// {8..15,24..31} lanes 16-31 (ISA 16-bit A layout).
__device__ __forceinline__ v16bf load_a16x32(const u16* base, int ld, int lane) {
    int m = lane & 15, hf = lane >> 4;
    const u16* p = base + (size_t)m * ld + hf * 8;
    FragU f;
    f.q[0] = *(const uint4*)(p);
    f.q[1] = *(const uint4*)(p + 16);
    return f.v;
}

// B fragment: 32x16 bf16, column n = lane&15; K-chunks {0..7,16..23} lanes 0-15,
// {8..15,24..31} lanes 16-31 (matches A layout). Storage "n-major, K contiguous".
__device__ __forceinline__ v16bf load_b32x16(const u16* base, int ld, int lane) {
    int n = lane & 15, hf = lane >> 4;
    const u16* p = base + (size_t)n * ld + hf * 8;
    FragU f;
    f.q[0] = *(const uint4*)(p);
    f.q[1] = *(const uint4*)(p + 16);
    return f.v;
}

__device__ __forceinline__ v8f wmma_bf16(v16bf a, v16bf b, v8f c) {
    return __builtin_amdgcn_wmma_f32_16x16x32_bf16(false, a, false, b, (short)0, c,
                                                   false, false);
}

// ---------------- Tensor Data Mover helpers ----------------
// LDS byte offset of a __shared__ object (generic -> addrspace(3) -> int)
__device__ __forceinline__ u32 lds_off(const void* p) {
    return (u32)(u64)(const __attribute__((address_space(3))) void*)p;
}

// Issue a 2D tensor_load_to_lds. D# per CDNA5 ISA:
//  group0: [1:0]=count=1, [63:32]=lds_addr, [120:64]=global byte addr, [127:126]=2
//  group1: [17:16]=data_size(1 => 2B), [79:48]=tensor_dim0, [111:80]=tensor_dim1,
//          [127:112]=tile_dim0, [143:128]=tile_dim1, [207:160]=tensor_dim0_stride
// tile_dim2 = 0 (2D); groups 2/3 passed as zero quads (higher dims unused).
__device__ __forceinline__ void tdm_load_2d(u32 ldsa, u64 gaddr, u32 tile0, u32 tile1,
                                            u64 stride0, u32 tdim0, u32 tdim1) {
    u32x4 g0;
    g0.x = 1u;                                    // count = 1
    g0.y = (u32)__builtin_amdgcn_readfirstlane((int)ldsa);
    g0.z = (u32)__builtin_amdgcn_readfirstlane((int)(u32)gaddr);
    g0.w = (u32)__builtin_amdgcn_readfirstlane(
               (int)(((u32)(gaddr >> 32) & 0x01FFFFFFu) | 0x80000000u)); // type=2
    u32x8 g1;
    g1[0] = 0x10000u;                             // data_size = 2 bytes
    g1[1] = (tdim0 & 0xFFFFu) << 16;
    g1[2] = (tdim0 >> 16) | ((tdim1 & 0xFFFFu) << 16);
    g1[3] = (tdim1 >> 16) | (tile0 << 16);
    g1[4] = tile1;                                // tile_dim1; tile_dim2 = 0
    g1[5] = (u32)stride0;
    g1[6] = (u32)(stride0 >> 32) & 0xFFFFu;       // tensor_dim1_stride = 0
    g1[7] = 0u;
    u32x4 z = {0u, 0u, 0u, 0u};
    asm volatile("tensor_load_to_lds %0, %1, %2, %3"
                 :
                 : "s"(g0), "s"(g1), "s"(z), "s"(z)
                 : "memory");
}

// ---------------- pack kernels ----------------
__global__ void k_pack_x(const float* __restrict__ x, u16* __restrict__ Xp) {
    size_t i = (size_t)blockIdx.x * 256 + threadIdx.x; // B*NPAD*EMBED threads
    int c = (int)(i % EMBED);
    size_t r = i / EMBED;
    int b = (int)(r / NPAD);
    int nr = (int)(r % NPAD);
    Xp[i] = (nr < NTOK) ? f2bf(x[((size_t)b * NTOK + nr) * EMBED + c]) : (u16)0;
}

__global__ void k_pack_w(const float* __restrict__ src, u16* __restrict__ dst, long n) {
    size_t i = (size_t)blockIdx.x * 256 + threadIdx.x;
    if (i < (size_t)n) dst[i] = f2bf(src[i]);
}

// ---------------- QKV projection GEMM ----------------
// 32x64 register-blocked tile per wave: 8 WMMA per 12 b128 loads.
// C[6208,2304] = Xp @ qkv_w^T; scatter into Q/K [bh][NPAD][64], Vt [bh][64][NPAD].
__global__ void k_qkv_gemm(const u16* __restrict__ Xp, const u16* __restrict__ Wq,
                           u16* __restrict__ Qb, u16* __restrict__ Kb,
                           u16* __restrict__ Vt) {
    int wave = threadIdx.x >> 5, lane = threadIdx.x & 31;
    int tile = blockIdx.x * 8 + wave;          // (MROWS/32)*(QKVDIM/64) = 194*36
    int tn = tile % (QKVDIM / 64);             // 0..35
    int tm = tile / (QKVDIM / 64);             // 0..193
    const u16* arow = Xp + (size_t)tm * 32 * EMBED;
    const u16* brow = Wq + (size_t)tn * 64 * EMBED;
    v8f acc[2][4] = {};
    for (int k0 = 0; k0 < EMBED; k0 += 32) {
        v16bf a0 = load_a16x32(arow + k0, EMBED, lane);
        v16bf a1 = load_a16x32(arow + (size_t)16 * EMBED + k0, EMBED, lane);
#pragma unroll
        for (int ni = 0; ni < 4; ++ni) {
            v16bf b = load_b32x16(brow + (size_t)ni * 16 * EMBED + k0, EMBED, lane);
            acc[0][ni] = wmma_bf16(a0, b, acc[0][ni]);
            acc[1][ni] = wmma_bf16(a1, b, acc[1][ni]);
        }
    }
    int hf = lane >> 4, n16 = lane & 15;
    int col_base = tn * 64;
    int comp = col_base / EMBED;               // uniform per wave
    int h = (col_base % EMBED) >> 6;           // uniform per wave
#pragma unroll
    for (int mi = 0; mi < 2; ++mi)
#pragma unroll
        for (int ni = 0; ni < 4; ++ni) {
            int dh = ni * 16 + n16;
#pragma unroll
            for (int i = 0; i < 8; ++i) {
                int row = tm * 32 + mi * 16 + i + hf * 8;
                int b_ = row / NPAD, nr = row % NPAD;
                size_t bh = (size_t)(b_ * NHEAD + h);
                u16 v = f2bf(acc[mi][ni][i]);
                if (comp == 0)
                    Qb[(bh * NPAD + nr) * HDIM + dh] = v;
                else if (comp == 1)
                    Kb[(bh * NPAD + nr) * HDIM + dh] = v;
                else
                    Vt[(bh * HDIM + dh) * NPAD + nr] = v;
            }
        }
}

// ---------------- flash attention ----------------
// blockIdx.y = (b,h); 8 waves = 8 consecutive query tiles sharing K/V.
// K tile (32x64, contiguous 4KB) and V tile (64 dh x 32 keys, stride NPAD) are
// TDM-loaded into double-buffered LDS by wave 0; s_wait_tensorcnt + barrier.
__global__ void k_attn(const u16* __restrict__ Qb, const u16* __restrict__ Kb,
                       const u16* __restrict__ Vt, float* __restrict__ Obuf) {
    __shared__ __align__(16) u16 sK[2][32 * 64]; // 2 x 4KB
    __shared__ __align__(16) u16 sV[2][64 * 32]; // 2 x 4KB
    __shared__ __align__(16) u16 sP[8][16 * 32]; // per-wave P tile
    int wave = threadIdx.x >> 5, lane = threadIdx.x & 31;
    int bh = blockIdx.y;                       // 0..23
    int qt = blockIdx.x * 8 + wave;
    if (qt >= NQT) qt = NQT - 1;               // tail waves duplicate (benign)
    const u16* Qp = Qb + (size_t)bh * NPAD * HDIM;
    const u16* Kp = Kb + (size_t)bh * NPAD * HDIM;
    const u16* Vp = Vt + (size_t)bh * HDIM * NPAD;

    v16bf aq0 = load_a16x32(Qp + (size_t)qt * 16 * HDIM + 0, HDIM, lane);
    v16bf aq1 = load_a16x32(Qp + (size_t)qt * 16 * HDIM + 32, HDIM, lane);

    v8f o0 = {}, o1 = {}, o2 = {}, o3 = {};
    float mrow[8], lrow[8];
#pragma unroll
    for (int i = 0; i < 8; ++i) {
        mrow[i] = -1e30f;
        lrow[i] = 0.f;
    }
    u16* myl = sP[wave];
    int hf = lane >> 4, n16 = lane & 15;

    // prologue: stage key-block 0 into buffer 0
    if (wave == 0) {
        tdm_load_2d(lds_off(sK[0]), (u64)(size_t)Kp, 2048u, 1u, 2048u,
                    (u32)NPAD * HDIM, 1u);
        tdm_load_2d(lds_off(sV[0]), (u64)(size_t)Vp, 32u, 64u, (u64)NPAD,
                    (u32)NPAD, (u32)HDIM);
        __builtin_amdgcn_s_wait_tensorcnt(0);
    }
    __syncthreads();

    int cur = 0;
    for (int jb = 0; jb < NKB; ++jb) {
        // stage next key block into the other buffer while computing this one
        if (wave == 0 && jb + 1 < NKB) {
            tdm_load_2d(lds_off(sK[cur ^ 1]),
                        (u64)(size_t)(Kp + (size_t)(jb + 1) * 32 * HDIM), 2048u, 1u,
                        2048u, (u32)NPAD * HDIM, 1u);
            tdm_load_2d(lds_off(sV[cur ^ 1]),
                        (u64)(size_t)(Vp + (size_t)(jb + 1) * 32), 32u, 64u, (u64)NPAD,
                        (u32)NPAD, (u32)HDIM);
        }
        const u16* kT = sK[cur];
        const u16* vT = sV[cur];

        v8f s0 = {}, s1 = {};
        s0 = wmma_bf16(aq0, load_b32x16(kT, HDIM, lane), s0);
        s0 = wmma_bf16(aq1, load_b32x16(kT + 32, HDIM, lane), s0);
        s1 = wmma_bf16(aq0, load_b32x16(kT + 16 * HDIM, HDIM, lane), s1);
        s1 = wmma_bf16(aq1, load_b32x16(kT + 16 * HDIM + 32, HDIM, lane), s1);

        // scale + pad-key masking (column index = lane&15 in C layout)
        int key0 = jb * 32 + n16, key1 = jb * 32 + 16 + n16;
        bool in0 = key0 < NTOK, in1 = key1 < NTOK;
#pragma unroll
        for (int i = 0; i < 8; ++i) {
            s0[i] = in0 ? s0[i] * SCALE : -1e30f;
            s1[i] = in1 ? s1[i] * SCALE : -1e30f;
        }
        // online softmax: row = VGPR idx (+8 upper half); reduce over 16 lanes
#pragma unroll
        for (int i = 0; i < 8; ++i) {
            float v = fmaxf(s0[i], s1[i]);
            for (int off = 1; off < 16; off <<= 1) v = fmaxf(v, __shfl_xor(v, off, 32));
            float nm = fmaxf(mrow[i], v);
            float a = __expf(mrow[i] - nm);
            float p0 = __expf(s0[i] - nm);
            float p1 = __expf(s1[i] - nm);
            s0[i] = p0;
            s1[i] = p1;
            float rs = p0 + p1;
            for (int off = 1; off < 16; off <<= 1) rs += __shfl_xor(rs, off, 32);
            lrow[i] = lrow[i] * a + rs;
            mrow[i] = nm;
            o0[i] *= a;
            o1[i] *= a;
            o2[i] *= a;
            o3[i] *= a;
        }
        // P (16x32) -> per-wave LDS tile -> A fragment.
        // Same-wave LDS ops complete in order (DScnt): no barrier needed.
#pragma unroll
        for (int i = 0; i < 8; ++i) {
            int r = i + hf * 8;
            myl[r * 32 + n16] = f2bf(s0[i]);
            myl[r * 32 + 16 + n16] = f2bf(s1[i]);
        }
        v16bf pf = load_a16x32(myl, 32, lane);
        o0 = wmma_bf16(pf, load_b32x16(vT + 0 * 16 * 32, 32, lane), o0);
        o1 = wmma_bf16(pf, load_b32x16(vT + 1 * 16 * 32, 32, lane), o1);
        o2 = wmma_bf16(pf, load_b32x16(vT + 2 * 16 * 32, 32, lane), o2);
        o3 = wmma_bf16(pf, load_b32x16(vT + 3 * 16 * 32, 32, lane), o3);

        if (wave == 0) __builtin_amdgcn_s_wait_tensorcnt(0);
        __syncthreads();
        cur ^= 1;
    }
    int b = bh / NHEAD, h = bh % NHEAD;
#pragma unroll
    for (int i = 0; i < 8; ++i) {
        float inv = 1.f / lrow[i];
        int row = qt * 16 + i + hf * 8;
        float* dst = Obuf + ((size_t)b * NPAD + row) * EMBED + h * HDIM;
        dst[0 * 16 + n16] = o0[i] * inv;
        dst[1 * 16 + n16] = o1[i] * inv;
        dst[2 * 16 + n16] = o2[i] * inv;
        dst[3 * 16 + n16] = o3[i] * inv;
    }
}

// ---------------- depthwise 3x3 conv (local pos embed), added into Obuf ------
__global__ void k_convpe(const u16* __restrict__ Vt, const float* __restrict__ cw,
                         const float* __restrict__ cb, float* __restrict__ Obuf) {
    size_t i = (size_t)blockIdx.x * 256 + threadIdx.x; // B*H*64*3072
    int sp = (int)(i % 3072);
    size_t t = i / 3072;
    int c = (int)(t % HDIM);
    t /= HDIM;
    int h = (int)(t % NHEAD);
    int b = (int)(t / NHEAD);
    int r = sp / 48, wc = sp % 48;
    const u16* vrow = Vt + (((size_t)(b * NHEAD + h)) * HDIM + c) * NPAD + 1;
    const float* wk = cw + c * 9;
    float acc = cb[c];
#pragma unroll
    for (int dr = -1; dr <= 1; ++dr)
#pragma unroll
        for (int dc = -1; dc <= 1; ++dc) {
            int rr = r + dr, cc = wc + dc;
            if (rr >= 0 && rr < 64 && cc >= 0 && cc < 48)
                acc += bf2f(vrow[rr * 48 + cc]) * wk[(dr + 1) * 3 + (dc + 1)];
        }
    int n = 1 + sp;
    Obuf[((size_t)b * NPAD + n) * EMBED + h * HDIM + c] += acc;
}

// ---------------- output projection GEMM ----------------
// 32x64 register-blocked; out[B*NTOK,768] = Opk @ proj_w^T + proj_b
__global__ void k_proj_gemm(const u16* __restrict__ Opk, const u16* __restrict__ Wp,
                            const float* __restrict__ bias, float* __restrict__ out) {
    int wave = threadIdx.x >> 5, lane = threadIdx.x & 31;
    int tile = blockIdx.x * 8 + wave;          // 194 * 12
    int tn = tile % (EMBED / 64);              // 0..11
    int tm = tile / (EMBED / 64);              // 0..193
    const u16* arow = Opk + (size_t)tm * 32 * EMBED;
    const u16* brow = Wp + (size_t)tn * 64 * EMBED;
    v8f acc[2][4] = {};
    for (int k0 = 0; k0 < EMBED; k0 += 32) {
        v16bf a0 = load_a16x32(arow + k0, EMBED, lane);
        v16bf a1 = load_a16x32(arow + (size_t)16 * EMBED + k0, EMBED, lane);
#pragma unroll
        for (int ni = 0; ni < 4; ++ni) {
            v16bf b = load_b32x16(brow + (size_t)ni * 16 * EMBED + k0, EMBED, lane);
            acc[0][ni] = wmma_bf16(a0, b, acc[0][ni]);
            acc[1][ni] = wmma_bf16(a1, b, acc[1][ni]);
        }
    }
    int hf = lane >> 4, n16 = lane & 15;
#pragma unroll
    for (int ni = 0; ni < 4; ++ni) {
        int col = tn * 64 + ni * 16 + n16;
        float bb = bias[col];
#pragma unroll
        for (int mi = 0; mi < 2; ++mi)
#pragma unroll
            for (int i = 0; i < 8; ++i) {
                int row = tm * 32 + mi * 16 + i + hf * 8;
                int b_ = row / NPAD, nr = row % NPAD;
                if (nr < NTOK)
                    out[((size_t)b_ * NTOK + nr) * EMBED + col] = acc[mi][ni][i] + bb;
            }
    }
}

// ---------------- host launch ----------------
extern "C" void kernel_launch(void* const* d_in, const int* in_sizes, int n_in,
                              void* d_out, int out_size, void* d_ws, size_t ws_size,
                              hipStream_t stream) {
    const float* x = (const float*)d_in[0];
    const float* qkv_w = (const float*)d_in[1];
    const float* proj_w = (const float*)d_in[2];
    const float* proj_b = (const float*)d_in[3];
    const float* conv_w = (const float*)d_in[4];
    const float* conv_b = (const float*)d_in[5];
    float* out = (float*)d_out;

    char* ws = (char*)d_ws;
    size_t off = 0;
    auto alloc = [&](size_t bytes) -> char* {
        char* p = ws + off;
        off += (bytes + 255) & ~(size_t)255;
        return p;
    };
    u16* Xp = (u16*)alloc((size_t)MROWS * EMBED * 2);
    u16* Wq = (u16*)alloc((size_t)QKVDIM * EMBED * 2);
    u16* Wpj = (u16*)alloc((size_t)EMBED * EMBED * 2);
    u16* Qb = (u16*)alloc((size_t)BATCH * NHEAD * NPAD * HDIM * 2);
    u16* Kb = (u16*)alloc((size_t)BATCH * NHEAD * NPAD * HDIM * 2);
    u16* Vt = (u16*)alloc((size_t)BATCH * NHEAD * HDIM * NPAD * 2);
    float* Obuf = (float*)alloc((size_t)MROWS * EMBED * 4);
    u16* Opk = (u16*)alloc((size_t)MROWS * EMBED * 2);

    // 1) pack inputs to bf16
    k_pack_x<<<(MROWS * EMBED) / 256, 256, 0, stream>>>(x, Xp);
    k_pack_w<<<(QKVDIM * EMBED) / 256, 256, 0, stream>>>(qkv_w, Wq, (long)QKVDIM * EMBED);
    k_pack_w<<<(EMBED * EMBED) / 256, 256, 0, stream>>>(proj_w, Wpj, (long)EMBED * EMBED);

    // 2) QKV projection (WMMA, 32x64 tiles): 194*36 wave-tiles, 8 waves/block
    k_qkv_gemm<<<(MROWS / 32) * (QKVDIM / 64) / 8, 256, 0, stream>>>(Xp, Wq, Qb, Kb, Vt);

    // 3) flash attention (WMMA + TDM double-buffered K/V staging)
    dim3 agrid((NQT + 7) / 8, BATCH * NHEAD);
    k_attn<<<agrid, 256, 0, stream>>>(Qb, Kb, Vt, Obuf);

    // 4) local position embedding (depthwise conv) added into Obuf
    k_convpe<<<(BATCH * NHEAD * HDIM * 3072) / 256, 256, 0, stream>>>(Vt, conv_w, conv_b,
                                                                     Obuf);

    // 5) pack attention output, then projection GEMM (WMMA) + bias
    k_pack_w<<<(MROWS * EMBED) / 256, 256, 0, stream>>>(Obuf, Opk, (long)MROWS * EMBED);
    k_proj_gemm<<<(MROWS / 32) * (EMBED / 64) / 8, 256, 0, stream>>>(Opk, Wpj, proj_b,
                                                                    out);
}
